// TransformerOCModel_43404939493999
// MI455X (gfx1250) — matrix operand order repaired and checked by reading, not compiled, not run on hardware
//
#include <hip/hip_runtime.h>
#include <hip/hip_bf16.h>
#include <math.h>

typedef _Float16 h16;
typedef __attribute__((ext_vector_type(16))) _Float16 v16h;
typedef __attribute__((ext_vector_type(8)))  _Float16 v8h;
typedef __attribute__((ext_vector_type(8)))  float    v8f;

#define PI_F 3.14159265358979f

// ---------------------------------------------------------------------------
// Fragment helpers (contiguous k order k = 16*g + i, wave32)
// ---------------------------------------------------------------------------
__device__ __forceinline__ v16h load_afrag(const h16* p) {
    // p = rowBase + k0 + g*16 ; halves i -> K = k0 + g*16 + i (contiguous 16)
    v8h c0 = *(const v8h*)p;
    v8h c1 = *(const v8h*)(p + 8);
    v16h a;
#pragma unroll
    for (int i = 0; i < 8; ++i) { a[i] = c0[i]; a[8 + i] = c1[i]; }
    return a;
}

__device__ __forceinline__ v16h load_bfrag(const h16* p) {
    // p = colBase + k0 + g*16 ; halves i -> K = k0 + g*16 + i (contiguous 16)
    v8h c0 = *(const v8h*)p;
    v8h c1 = *(const v8h*)(p + 8);
    v16h b;
#pragma unroll
    for (int i = 0; i < 8; ++i) { b[i] = c0[i]; b[8 + i] = c1[i]; }
    return b;
}

__device__ __forceinline__ v8f wmma_f16(v16h a, v16h b, v8f c) {
    return __builtin_amdgcn_wmma_f32_16x16x32_f16(false, a, false, b, (short)0, c, false, false);
}

// ---------------------------------------------------------------------------
// Small utility kernels
// ---------------------------------------------------------------------------
__global__ void cvt_kernel(const float* __restrict__ in, h16* __restrict__ out, int cnt) {
    int i = blockIdx.x * 256 + threadIdx.x;
    if (i < cnt) out[i] = (h16)in[i];
}

__global__ void embed_kernel(const float* __restrict__ xr, const float* __restrict__ Win,
                             const float* __restrict__ bin, float* __restrict__ xf) {
    int idx = blockIdx.x * 256 + threadIdx.x;       // 2048*256
    int m = idx >> 8, nn = idx & 255;
    float acc = bin[nn];
#pragma unroll
    for (int k = 0; k < 8; ++k) acc += xr[m * 8 + k] * Win[nn * 8 + k];
    xf[idx] = acc;
}

__global__ void etaphi_kernel(const float* __restrict__ xr, float* __restrict__ eta,
                              float* __restrict__ phi) {
    int m = blockIdx.x * 256 + threadIdx.x;
    if (m < 2048) {
        eta[m] = xr[m * 8 + 1] * 5.24f - 2.62f;
        phi[m] = xr[m * 8 + 2] * 6.2832f - 3.1416f;
    }
}

// LayerNorm: one wave per row of 256, lane handles 8 contiguous cols
__global__ __launch_bounds__(256) void ln_kernel(const float* __restrict__ x,
                                                 const float* __restrict__ gg,
                                                 const float* __restrict__ bb,
                                                 h16* __restrict__ out) {
    int lane = threadIdx.x & 31, wid = threadIdx.x >> 5;
    int row = blockIdx.x * 8 + wid;
    const float4* p = (const float4*)(x + (size_t)row * 256 + lane * 8);
    float4 a = p[0], b4 = p[1];
    float v[8] = {a.x, a.y, a.z, a.w, b4.x, b4.y, b4.z, b4.w};
    float s = 0.f;
#pragma unroll
    for (int i = 0; i < 8; ++i) s += v[i];
#pragma unroll
    for (int off = 1; off < 32; off <<= 1) s += __shfl_xor(s, off, 32);
    float mean = s * (1.f / 256.f);
    float vs = 0.f;
#pragma unroll
    for (int i = 0; i < 8; ++i) { float d = v[i] - mean; vs += d * d; }
#pragma unroll
    for (int off = 1; off < 32; off <<= 1) vs += __shfl_xor(vs, off, 32);
    float rstd = rsqrtf(vs * (1.f / 256.f) + 1e-5f);
    v8h o;
#pragma unroll
    for (int i = 0; i < 8; ++i) {
        int col = lane * 8 + i;
        o[i] = (h16)((v[i] - mean) * rstd * gg[col] + bb[col]);
    }
    *(v8h*)(out + (size_t)row * 256 + lane * 8) = o;
}

__global__ void beta_kernel(const h16* __restrict__ r2, const h16* __restrict__ w,
                            const float* __restrict__ bb2, float* __restrict__ out) {
    int m = blockIdx.x * 256 + threadIdx.x;
    float acc = bb2[0];
#pragma unroll 8
    for (int k = 0; k < 128; ++k) acc += (float)r2[m * 128 + k] * (float)w[k];
    float b = 1.f / (1.f + __expf(-acc));
    out[m] = fminf(fmaxf(b, 1e-6f), 1.f - 1e-6f);
}

// ---------------------------------------------------------------------------
// Generic WMMA GEMM:  C[M=2048, N] = A[M,K](f16) @ Wt[N,K]^T(f16) + bias
// One wave computes 64 rows x 16 cols (4 M-tiles reuse one B fragment).
// ---------------------------------------------------------------------------
enum { MODE_QKV = 0, MODE_RELU16 = 1, MODE_RESID = 2, MODE_F32B = 3 };

template <int MODE>
__global__ __launch_bounds__(256) void gemm16k(const h16* __restrict__ A,
                                               const h16* __restrict__ Wt,
                                               const float* __restrict__ bias,
                                               h16* __restrict__ o16,
                                               h16* __restrict__ vtb,
                                               float* __restrict__ of32,
                                               int Ncols, int K, int nvalid) {
    const int wave = (blockIdx.x * blockDim.x + threadIdx.x) >> 5;
    const int lane = threadIdx.x & 31;
    const int n = lane & 15, g = lane >> 4;
    const int ntiles = Ncols >> 4;
    const int m0 = (wave / ntiles) * 64;
    const int n0 = (wave % ntiles) * 16;

    v8f cacc[4];
#pragma unroll
    for (int t = 0; t < 4; ++t) cacc[t] = (v8f){};

    const h16* wbase = Wt + (size_t)(n0 + n) * K + g * 16;
    for (int k0 = 0; k0 < K; k0 += 32) {
        v16h b = load_bfrag(wbase + k0);
#pragma unroll
        for (int t = 0; t < 4; ++t) {
            v16h a = load_afrag(A + (size_t)(m0 + t * 16 + n) * K + k0 + g * 16);
            cacc[t] = wmma_f16(a, b, cacc[t]);
        }
    }

    const int col = n0 + n;
    float bv;
    if (MODE == MODE_F32B) bv = (col < nvalid) ? bias[col] : 0.f;
    else                   bv = bias[col];

#pragma unroll
    for (int t = 0; t < 4; ++t) {
#pragma unroll
        for (int r = 0; r < 8; ++r) {
            int row = m0 + t * 16 + r + g * 8;
            float val = cacc[t][r] + bv;
            if (MODE == MODE_QKV) {
                if (n0 < 512) o16[(size_t)row * 512 + col] = (h16)val;            // Q,K
                else          vtb[(size_t)(col - 512) * 2048 + row] = (h16)val;   // V transposed
            } else if (MODE == MODE_RELU16) {
                o16[(size_t)row * Ncols + col] = (h16)fmaxf(val, 0.f);
            } else if (MODE == MODE_RESID) {
                size_t idx = (size_t)row * Ncols + col;
                of32[idx] = of32[idx] + val;                                      // residual add
            } else { // MODE_F32B
                if (col < nvalid) of32[(size_t)row * nvalid + col] = val;
            }
        }
    }
}

// ---------------------------------------------------------------------------
// Flash attention: one wave = 16 query rows of one head; online softmax.
// qkvh: [2048, 512] f16 (Q cols 0..255, K cols 256..511) ; vt: [256, 2048] f16
// ---------------------------------------------------------------------------
__device__ __forceinline__ float wrap_pi(float d) {
    d = (d > PI_F) ? d - 2.f * PI_F : d;
    d = (d < -PI_F) ? d + 2.f * PI_F : d;
    return d;
}

__global__ __launch_bounds__(256) void attn_kernel(const h16* __restrict__ qkvh,
                                                   const h16* __restrict__ vt,
                                                   const float* __restrict__ eta,
                                                   const float* __restrict__ phi,
                                                   h16* __restrict__ oh) {
    __shared__ __align__(32) _Float16 ptile[8][16][32];
    const int wid = threadIdx.x >> 5;
    const int lane = threadIdx.x & 31;
    const int n = lane & 15, g = lane >> 4;
    const int gw = blockIdx.x * 8 + wid;
    const int hh = gw & 7;
    const int m0 = (gw >> 3) << 4;
    const float scale = 0.17677669529663687f;  // 1/sqrt(32)

    // Q fragment (A layout), pre-scaled
    v16h qa = load_afrag(qkvh + (size_t)(m0 + n) * 512 + hh * 32 + g * 16);
#pragma unroll
    for (int i = 0; i < 16; ++i) qa[i] = qa[i] * (h16)scale;

    float er[8], pr[8];
#pragma unroll
    for (int r = 0; r < 8; ++r) {
        int m = m0 + r + g * 8;
        er[r] = eta[m]; pr[r] = phi[m];
    }

    v8f o0 = (v8f){}, o1 = (v8f){};
    float mi[8], li[8];
#pragma unroll
    for (int r = 0; r < 8; ++r) { mi[r] = -1e30f; li[r] = 0.f; }

    for (int j0 = 0; j0 < 2048; j0 += 32) {
        const h16* kp = qkvh + (size_t)(j0 + n) * 512 + 256 + hh * 32 + g * 16;
        v16h kb0 = load_bfrag(kp);
        v16h kb1 = load_bfrag(kp + 16 * 512);
        v8f z = (v8f){};
        v8f s0 = wmma_f16(qa, kb0, z);
        v8f s1 = wmma_f16(qa, kb1, z);

        float ek0 = eta[j0 + n],      pk0 = phi[j0 + n];
        float ek1 = eta[j0 + 16 + n], pk1 = phi[j0 + 16 + n];

#pragma unroll
        for (int r = 0; r < 8; ++r) {
            float d0 = er[r] - ek0, f0 = wrap_pi(pr[r] - pk0);
            float d1 = er[r] - ek1, f1 = wrap_pi(pr[r] - pk1);
            float v0 = (d0 * d0 + f0 * f0 > 0.04f) ? -1e30f : s0[r];
            float v1 = (d1 * d1 + f1 * f1 > 0.04f) ? -1e30f : s1[r];
            float bm = fmaxf(v0, v1);
#pragma unroll
            for (int off = 1; off < 16; off <<= 1) bm = fmaxf(bm, __shfl_xor(bm, off, 32));
            float nm = fmaxf(mi[r], bm);
            float corr = __expf(mi[r] - nm);
            float p0 = __expf(v0 - nm);
            float p1 = __expf(v1 - nm);
            float rs = p0 + p1;
#pragma unroll
            for (int off = 1; off < 16; off <<= 1) rs += __shfl_xor(rs, off, 32);
            li[r] = li[r] * corr + rs;
            mi[r] = nm;
            o0[r] *= corr; o1[r] *= corr;
            ptile[wid][r + g * 8][n] = (h16)p0;
            ptile[wid][r + g * 8][16 + n] = (h16)p1;
        }
        asm volatile("s_wait_dscnt 0" ::: "memory");  // wave-private LDS bounce complete

        // P in A-frag layout straight from LDS
        const _Float16* pp = &ptile[wid][n][g * 16];
        v16h pa = load_afrag(pp);

        const h16* vp = vt + (size_t)(hh * 32 + n) * 2048 + j0 + g * 16;
        v16h vb0 = load_bfrag(vp);
        v16h vb1 = load_bfrag(vp + 16 * 2048);
        o0 = wmma_f16(pa, vb0, o0);
        o1 = wmma_f16(pa, vb1, o1);
    }

#pragma unroll
    for (int r = 0; r < 8; ++r) {
        float inv = 1.f / li[r];
        size_t row = (size_t)(m0 + r + g * 8);
        oh[row * 256 + hh * 32 + n]      = (h16)(o0[r] * inv);
        oh[row * 256 + hh * 32 + 16 + n] = (h16)(o1[r] * inv);
    }
}

// ---------------------------------------------------------------------------
// Host orchestration
// ---------------------------------------------------------------------------
extern "C" void kernel_launch(void* const* d_in, const int* in_sizes, int n_in,
                              void* d_out, int out_size, void* d_ws, size_t ws_size,
                              hipStream_t stream) {
    (void)in_sizes; (void)n_in; (void)out_size; (void)ws_size;
    constexpr int N = 2048, H = 256, FF = 512, L = 6;

    const float* x_raw = (const float*)d_in[0];
    const float* Win  = (const float*)d_in[2];
    const float* b_in = (const float*)d_in[3];
    const float* Wqkv = (const float*)d_in[4];
    const float* bqkv = (const float*)d_in[5];
    const float* Wo   = (const float*)d_in[6];
    const float* bo   = (const float*)d_in[7];
    const float* W1   = (const float*)d_in[8];
    const float* b1   = (const float*)d_in[9];
    const float* W2   = (const float*)d_in[10];
    const float* b2   = (const float*)d_in[11];
    const float* g1   = (const float*)d_in[12];
    const float* be1  = (const float*)d_in[13];
    const float* g2   = (const float*)d_in[14];
    const float* be2  = (const float*)d_in[15];
    const float* Wl1  = (const float*)d_in[16];
    const float* bl1  = (const float*)d_in[17];
    const float* Wl2  = (const float*)d_in[18];
    const float* bl2  = (const float*)d_in[19];
    const float* Wb1  = (const float*)d_in[20];
    const float* bb1  = (const float*)d_in[21];
    const float* Wb2  = (const float*)d_in[22];
    const float* bb2  = (const float*)d_in[23];
    float* out = (float*)d_out;

    char* ws = (char*)d_ws;
    size_t off = 0;
    auto alloc = [&](size_t bytes) -> void* {
        void* p = ws + off;
        off += (bytes + 255) & ~(size_t)255;
        return p;
    };

    float* xf   = (float*)alloc((size_t)N * H * 4);
    h16* hln    = (h16*)alloc((size_t)N * H * 2);
    h16* qkvh   = (h16*)alloc((size_t)N * 512 * 2);
    h16* vtb    = (h16*)alloc((size_t)H * N * 2);
    h16* oh     = (h16*)alloc((size_t)N * H * 2);
    h16* t16    = (h16*)alloc((size_t)N * FF * 2);
    h16* r1     = (h16*)alloc((size_t)N * H * 2);
    h16* r2     = (h16*)alloc((size_t)N * 128 * 2);
    float* etaA = (float*)alloc((size_t)N * 4);
    float* phiA = (float*)alloc((size_t)N * 4);
    h16* wq16   = (h16*)alloc((size_t)L * 768 * H * 2);
    h16* wo16   = (h16*)alloc((size_t)L * H * H * 2);
    h16* w116   = (h16*)alloc((size_t)L * FF * H * 2);
    h16* w216   = (h16*)alloc((size_t)L * H * FF * 2);
    h16* wl116  = (h16*)alloc((size_t)H * H * 2);
    h16* wl216  = (h16*)alloc((size_t)32 * H * 2);   // padded 24 -> 32 rows
    h16* wb116  = (h16*)alloc((size_t)128 * H * 2);
    h16* wb216  = (h16*)alloc((size_t)128 * 2);

    auto cvt = [&](const float* src, h16* dst, int cnt) {
        cvt_kernel<<<(cnt + 255) / 256, 256, 0, stream>>>(src, dst, cnt);
    };
    cvt(Wqkv, wq16, L * 768 * H);
    cvt(Wo,   wo16, L * H * H);
    cvt(W1,   w116, L * FF * H);
    cvt(W2,   w216, L * H * FF);
    cvt(Wl1,  wl116, H * H);
    cvt(Wl2,  wl216, 24 * H);
    cvt(Wb1,  wb116, 128 * H);
    cvt(Wb2,  wb216, 128);

    embed_kernel<<<N * H / 256, 256, 0, stream>>>(x_raw, Win, b_in, xf);
    etaphi_kernel<<<8, 256, 0, stream>>>(x_raw, etaA, phiA);

    for (int l = 0; l < L; ++l) {
        ln_kernel<<<N / 8, 256, 0, stream>>>(xf, g1 + l * H, be1 + l * H, hln);
        // QKV: [2048,256] x [768,256]^T
        gemm16k<MODE_QKV><<<192, 256, 0, stream>>>(hln, wq16 + (size_t)l * 768 * H,
                                                   bqkv + l * 768, qkvh, vtb, nullptr,
                                                   768, H, 0);
        attn_kernel<<<128, 256, 0, stream>>>(qkvh, vtb, etaA, phiA, oh);
        // x += O @ Wo^T + bo
        gemm16k<MODE_RESID><<<64, 256, 0, stream>>>(oh, wo16 + (size_t)l * H * H,
                                                    bo + l * H, nullptr, nullptr, xf,
                                                    H, H, 0);
        ln_kernel<<<N / 8, 256, 0, stream>>>(xf, g2 + l * H, be2 + l * H, hln);
        // t = relu(h2 @ W1^T + b1)
        gemm16k<MODE_RELU16><<<128, 256, 0, stream>>>(hln, w116 + (size_t)l * FF * H,
                                                      b1 + l * FF, t16, nullptr, nullptr,
                                                      FF, H, 0);
        // x += t @ W2^T + b2
        gemm16k<MODE_RESID><<<64, 256, 0, stream>>>(t16, w216 + (size_t)l * H * FF,
                                                    b2 + l * H, nullptr, nullptr, xf,
                                                    H, FF, 0);
    }

    // heads
    cvt(xf, hln, N * H);
    gemm16k<MODE_RELU16><<<64, 256, 0, stream>>>(hln, wl116, bl1, r1, nullptr, nullptr,
                                                 H, H, 0);
    gemm16k<MODE_F32B><<<8, 256, 0, stream>>>(r1, wl216, bl2, nullptr, nullptr,
                                              out + N, 32, H, 24);   // coords [2048,24]
    gemm16k<MODE_RELU16><<<32, 256, 0, stream>>>(hln, wb116, bb1, r2, nullptr, nullptr,
                                                 128, H, 0);
    beta_kernel<<<8, 256, 0, stream>>>(r2, wb216, bb2, out);          // beta [2048,1]
}
